// TwoTowerModel_66013647339814
// MI455X (gfx1250) — matrix operand order repaired and checked by reading, not compiled, not run on hardware
//
#include <hip/hip_runtime.h>
#include <cstdint>
#include <cstddef>

// ---------------------------------------------------------------------------
// TwoTower hetero-GNN for MI455X (gfx1250, wave32, WMMA).
//
// Pipeline (all on `stream`, graph-capture safe):
//   1) pack weights -> bf16, column-major (WMMA-B-fragment friendly), fusing
//      [enc_w ; t_w] (K=192) and [wl ; wr] (K=256) stacks.
//   2) encoder GEMM: A=[x | posenc(rel_t)] (built in LDS, bf16), K=192.
//   3) per layer: zero agg, wave-per-edge scatter (native global_atomic_add_f32
//      via inline asm, device scope), fused GEMM (K=256) + bias + LayerNorm +
//      ReLU, in-place node update.
//   4) copy xu[:1024] to d_out.
// ---------------------------------------------------------------------------

typedef __attribute__((ext_vector_type(16))) __bf16 v16bf;
typedef __attribute__((ext_vector_type(8)))  float  v8f;

union V16 { v16bf v; uint4 u4[2]; };

#define CH     128
#define NUSER  50000
#define NITEM  50000
#define NEDGE  500000
#define ROWS_PER_BLOCK 64

// -ln(10000)/64  (pos-enc frequency exponent step)
#define NEG_LOG1E4_OVER_64 (-0.14391157f)

// Native CDNA5 float atomic add, no-return form, device scope (RMW at L2
// atomic units, visible to all WGPs; tracked by STOREcnt, drained at endpgm).
__device__ inline void gatomic_fadd(float* p, float v) {
    asm volatile("global_atomic_add_f32 %0, %1, off scope:SCOPE_DEV"
                 :: "v"((uint64_t)(uintptr_t)p), "v"(v)
                 : "memory");
}

// ---------------------------------------------------------------------------
// Wave-level GEMM core: one wave computes a 16x64 slice of a 64x128 tile.
// A tile lives in LDS as bf16 (row stride RS elements, 16B aligned).
// Wt is a KTOT x 128 weight matrix stored column-major bf16: Wt[n*KTOT + k].
// Fragment layouts follow CDNA5 ISA 7.12.2 (16-bit A 16x32, 16-bit B 32x16).
// ---------------------------------------------------------------------------
template <int KTOT, int RS>
__device__ inline void wave_gemm(const __bf16* a_lds,
                                 const __bf16* __restrict__ Wt,
                                 int lane, int rb, int chf, v8f acc[4]) {
    const int m     = rb * 16 + (lane & 15);
    const int koffA = (lane < 16) ? 0 : 8;    // A: lanes>=16 hold K 8..15 / 24..31
    const int koffB = (lane < 16) ? 0 : 8;    // B: lanes>=16 hold K 8..15 / 24..31
    const int nbase = chf * 64 + (lane & 15);
#pragma unroll
    for (int ks = 0; ks < KTOT; ks += 32) {
        V16 a;
        a.u4[0] = *reinterpret_cast<const uint4*>(a_lds + m * RS + ks + koffA);
        a.u4[1] = *reinterpret_cast<const uint4*>(a_lds + m * RS + ks + 16 + koffA);
#pragma unroll
        for (int nt = 0; nt < 4; ++nt) {
            V16 b;
            const __bf16* bp = Wt + (nbase + nt * 16) * KTOT + ks + koffB;
            b.u4[0] = *reinterpret_cast<const uint4*>(bp);
            b.u4[1] = *reinterpret_cast<const uint4*>(bp + 16);
            acc[nt] = __builtin_amdgcn_wmma_f32_16x16x32_bf16(
                false, a.v, false, b.v, (short)0, acc[nt], false, false);
        }
    }
}

// ---------------------------------------------------------------------------
// Weight pack: dst[n*(K0+K1)+k] = bf16( k<K0 ? w0[k][n] : w1[k-K0][n] )
// ---------------------------------------------------------------------------
__global__ __launch_bounds__(256) void pack_weights_kernel(
    const float* __restrict__ w0, const float* __restrict__ w1,
    int K0, int K1, __bf16* __restrict__ dst) {
    const int KT  = K0 + K1;
    const int idx = blockIdx.x * 256 + threadIdx.x;
    if (idx >= CH * KT) return;
    const int n = idx / KT, k = idx - n * KT;
    const float v = (k < K0) ? w0[k * CH + n] : w1[(k - K0) * CH + n];
    dst[n * KT + k] = (__bf16)v;
}

// ---------------------------------------------------------------------------
// Encoder: out[row] = x[row] @ enc_w + enc_b + posenc(rel_t) @ t_w + t_b
// Fused as [x | pe] (K=192) GEMM with pe built on the fly into LDS.
// ---------------------------------------------------------------------------
__global__ __launch_bounds__(256) void encoder_kernel(
    const float* __restrict__ x,      // M x 64
    const float* __restrict__ tnode,  // M
    const float* __restrict__ seed,   // B seeds
    const int*   __restrict__ batch,  // M
    const __bf16* __restrict__ Wt,    // 128 x 192 packed col-major
    const float* __restrict__ b_enc, const float* __restrict__ b_t,
    float* __restrict__ out, int M) {
    constexpr int KTOT = 192, RS = 200;  // RS multiple of 8 -> 16B aligned rows
    __shared__ __bf16 a_lds[ROWS_PER_BLOCK * RS];
    const int t = threadIdx.x;
    const int row0 = blockIdx.x * ROWS_PER_BLOCK;

    // Phase 1: build A tile (64 x 192 bf16): [x(64) | posenc(128)]
    {
        const int r = t >> 2, q = t & 3;  // 4 threads per row, 48 cols each
        const int row = row0 + r;
        const bool valid = row < M;
        float rel = 0.f;
        if (valid) rel = (seed[batch[row]] - tnode[row]) * (1.0f / 86400.0f);
        const int c0 = q * 48;
        for (int c = c0; c < c0 + 48; ++c) {
            float v = 0.f;
            if (valid) {
                if (c < 64) {
                    v = x[row * 64 + c];
                } else {
                    const int j = c - 64;
                    const float div = __expf((float)(j >> 1) * NEG_LOG1E4_OVER_64);
                    const float ang = rel * div;
                    v = (j & 1) ? __cosf(ang) : __sinf(ang);
                }
            }
            a_lds[r * RS + c] = (__bf16)v;
        }
    }
    __syncthreads();

    // Phase 2: WMMA GEMM
    const int lane = t & 31, wave = t >> 5;
    const int rb = wave >> 1, chf = wave & 1;
    v8f acc[4] = {};
    wave_gemm<KTOT, RS>(a_lds, Wt, lane, rb, chf, acc);

    // Epilogue: add combined bias, store f32
    const int baseRow = rb * 16 + ((lane >> 4) << 3);
#pragma unroll
    for (int nt = 0; nt < 4; ++nt) {
        const int col = chf * 64 + nt * 16 + (lane & 15);
        const float bb = b_enc[col] + b_t[col];
#pragma unroll
        for (int r = 0; r < 8; ++r) {
            const int row = row0 + baseRow + r;
            if (row < M) out[row * CH + col] = acc[nt][r] + bb;
        }
    }
}

// ---------------------------------------------------------------------------
// Edge scatter: agg[dst] += x[src]; one wave32 per edge, lane handles 4
// contiguous columns: one global_load_b128 gather + 4 native f32 atomics
// to consecutive dwords (same L2 line). Wave covers the full 512B row.
// edges layout: edges[0..nE) = src, edges[nE..2nE) = dst.
// ---------------------------------------------------------------------------
__global__ __launch_bounds__(256) void scatter_kernel(
    const float* __restrict__ x, const int* __restrict__ edges,
    float* __restrict__ agg, int nE) {
    const int gid = blockIdx.x * 256 + threadIdx.x;
    const int e = gid >> 5;
    if (e >= nE) return;
    const int lane = gid & 31;
    const int src = edges[e];
    const int dst = edges[nE + e];
    const float4 xv =
        *reinterpret_cast<const float4*>(x + (size_t)src * CH + lane * 4);
    float* ad = agg + (size_t)dst * CH + lane * 4;
    gatomic_fadd(ad + 0, xv.x);
    gatomic_fadd(ad + 1, xv.y);
    gatomic_fadd(ad + 2, xv.z);
    gatomic_fadd(ad + 3, xv.w);
}

// ---------------------------------------------------------------------------
// Layer: x[row] = relu(LN(agg[row]@Wl + b + x[row]@Wr)); fused K=256 GEMM,
// GEMM result staged into LDS (aliased on A tile) for the LayerNorm pass.
// In-place update is safe: each block reads only its own rows before writing.
// ---------------------------------------------------------------------------
__global__ __launch_bounds__(256) void layer_kernel(
    const float* __restrict__ agg, float* __restrict__ x,
    const __bf16* __restrict__ Wt, const float* __restrict__ bias,
    const float* __restrict__ g, const float* __restrict__ beta, int M) {
    constexpr int KTOT = 256, RS = 264, RSC = 136;
    __shared__ __align__(16) char smem[ROWS_PER_BLOCK * RSC * 4];  // 34816 B
    __bf16* a_lds = reinterpret_cast<__bf16*>(smem);               // 33792 B
    float*  c_lds = reinterpret_cast<float*>(smem);
    const int t = threadIdx.x;
    const int row0 = blockIdx.x * ROWS_PER_BLOCK;

    // Phase 1: A tile = [agg(128) | x(128)] as bf16
    {
        const int r = t >> 2, q = t & 3;
        const int row = row0 + r;
        const bool valid = row < M;
        const int c0 = q * 64;
        for (int c = c0; c < c0 + 64; ++c) {
            float v = 0.f;
            if (valid) v = (c < CH) ? agg[row * CH + c] : x[row * CH + (c - CH)];
            a_lds[r * RS + c] = (__bf16)v;
        }
    }
    __syncthreads();

    // Phase 2: WMMA GEMM
    const int lane = t & 31, wave = t >> 5;
    const int rb = wave >> 1, chf = wave & 1;
    v8f acc[4] = {};
    wave_gemm<KTOT, RS>(a_lds, Wt, lane, rb, chf, acc);
    __syncthreads();  // all waves done with a_lds before aliasing as c_lds

    // Phase 3: deposit GEMM result + bias into LDS (f32)
    const int baseRow = rb * 16 + ((lane >> 4) << 3);
#pragma unroll
    for (int nt = 0; nt < 4; ++nt) {
        const int col = chf * 64 + nt * 16 + (lane & 15);
        const float bb = bias[col];
#pragma unroll
        for (int r = 0; r < 8; ++r)
            c_lds[(baseRow + r) * RSC + col] = acc[nt][r] + bb;
    }
    __syncthreads();

    // Phase 4: LayerNorm + ReLU, one thread per row, in-place store
    if (t < ROWS_PER_BLOCK) {
        const int row = row0 + t;
        if (row < M) {
            const float* cr = c_lds + t * RSC;
            float mean = 0.f;
            for (int c = 0; c < CH; ++c) mean += cr[c];
            mean *= (1.0f / CH);
            float var = 0.f;
            for (int c = 0; c < CH; ++c) { const float d = cr[c] - mean; var += d * d; }
            var *= (1.0f / CH);
            const float inv = rsqrtf(var + 1e-5f);
            float* xr = x + (size_t)row * CH;
            for (int c = 0; c < CH; ++c) {
                const float y = (cr[c] - mean) * inv * g[c] + beta[c];
                xr[c] = fmaxf(y, 0.f);
            }
        }
    }
}

__global__ __launch_bounds__(256) void zero4_kernel(float4* __restrict__ p, int n4) {
    const int i = blockIdx.x * 256 + threadIdx.x;
    if (i < n4) p[i] = float4{0.f, 0.f, 0.f, 0.f};
}

__global__ __launch_bounds__(256) void copy4_kernel(const float4* __restrict__ s,
                                                    float4* __restrict__ d, int n4) {
    const int i = blockIdx.x * 256 + threadIdx.x;
    if (i < n4) d[i] = s[i];
}

// ---------------------------------------------------------------------------
extern "C" void kernel_launch(void* const* d_in, const int* in_sizes, int n_in,
                              void* d_out, int out_size, void* d_ws, size_t ws_size,
                              hipStream_t stream) {
    (void)in_sizes; (void)n_in; (void)out_size; (void)ws_size;

    const float* x_user    = (const float*)d_in[0];
    const float* x_item    = (const float*)d_in[1];
    const float* time_user = (const float*)d_in[2];
    const float* time_item = (const float*)d_in[3];
    const float* seed_time = (const float*)d_in[4];
    const int*   batch_user = (const int*)d_in[5];
    const int*   batch_item = (const int*)d_in[6];
    const int*   edge_u2i   = (const int*)d_in[7];
    const int*   edge_i2u   = (const int*)d_in[8];
    // params, insertion order of setup_inputs()
    const float* enc_user_w = (const float*)d_in[9];
    const float* enc_user_b = (const float*)d_in[10];
    const float* enc_item_w = (const float*)d_in[11];
    const float* enc_item_b = (const float*)d_in[12];
    const float* t_user_w   = (const float*)d_in[13];
    const float* t_user_b   = (const float*)d_in[14];
    const float* t_item_w   = (const float*)d_in[15];
    const float* t_item_b   = (const float*)d_in[16];
    // layers: base 17 + 10*l: u2i_wl, u2i_wr, u2i_b, i2u_wl, i2u_wr, i2u_b,
    //                         ln_user_g, ln_user_b, ln_item_g, ln_item_b

    // ---- workspace carve-up ----
    char* ws = (char*)d_ws;
    size_t off = 0;
    auto alloc = [&](size_t bytes) {
        char* p = ws + off;
        off = (off + bytes + 255) & ~(size_t)255;
        return p;
    };
    float* xu    = (float*)alloc((size_t)NUSER * CH * 4);
    float* xi    = (float*)alloc((size_t)NITEM * CH * 4);
    float* agg_i = (float*)alloc((size_t)NITEM * CH * 4);   // contiguous with agg_u
    float* agg_u = (float*)alloc((size_t)NUSER * CH * 4);
    __bf16* Wenc_u = (__bf16*)alloc((size_t)CH * 192 * 2);
    __bf16* Wenc_i = (__bf16*)alloc((size_t)CH * 192 * 2);
    __bf16* WiL[2], *WuL[2];
    for (int l = 0; l < 2; ++l) {
        WiL[l] = (__bf16*)alloc((size_t)CH * 256 * 2);
        WuL[l] = (__bf16*)alloc((size_t)CH * 256 * 2);
    }

    // ---- 1) pack weights ----
    {
        const int ne192 = CH * 192, ne256 = CH * 256;
        pack_weights_kernel<<<(ne192 + 255) / 256, 256, 0, stream>>>(
            enc_user_w, t_user_w, 64, 128, Wenc_u);
        pack_weights_kernel<<<(ne192 + 255) / 256, 256, 0, stream>>>(
            enc_item_w, t_item_w, 64, 128, Wenc_i);
        for (int l = 0; l < 2; ++l) {
            const float* u2i_wl = (const float*)d_in[17 + 10 * l + 0];
            const float* u2i_wr = (const float*)d_in[17 + 10 * l + 1];
            const float* i2u_wl = (const float*)d_in[17 + 10 * l + 3];
            const float* i2u_wr = (const float*)d_in[17 + 10 * l + 4];
            pack_weights_kernel<<<(ne256 + 255) / 256, 256, 0, stream>>>(
                u2i_wl, u2i_wr, 128, 128, WiL[l]);
            pack_weights_kernel<<<(ne256 + 255) / 256, 256, 0, stream>>>(
                i2u_wl, i2u_wr, 128, 128, WuL[l]);
        }
    }

    // ---- 2) encoders (fused feature + temporal PE GEMM) ----
    const int gU = (NUSER + ROWS_PER_BLOCK - 1) / ROWS_PER_BLOCK;
    const int gI = (NITEM + ROWS_PER_BLOCK - 1) / ROWS_PER_BLOCK;
    encoder_kernel<<<gU, 256, 0, stream>>>(x_user, time_user, seed_time, batch_user,
                                           Wenc_u, enc_user_b, t_user_b, xu, NUSER);
    encoder_kernel<<<gI, 256, 0, stream>>>(x_item, time_item, seed_time, batch_item,
                                           Wenc_i, enc_item_b, t_item_b, xi, NITEM);

    // ---- 3) GNN layers ----
    const int nAgg4 = (NUSER + NITEM) * CH / 4;  // agg_i ++ agg_u contiguous
    const int gScat = (NEDGE * 32 + 255) / 256;  // wave per edge
    for (int l = 0; l < 2; ++l) {
        const float* u2i_b     = (const float*)d_in[17 + 10 * l + 2];
        const float* i2u_b     = (const float*)d_in[17 + 10 * l + 5];
        const float* ln_user_g = (const float*)d_in[17 + 10 * l + 6];
        const float* ln_user_b = (const float*)d_in[17 + 10 * l + 7];
        const float* ln_item_g = (const float*)d_in[17 + 10 * l + 8];
        const float* ln_item_b = (const float*)d_in[17 + 10 * l + 9];

        zero4_kernel<<<(nAgg4 + 255) / 256, 256, 0, stream>>>((float4*)agg_i, nAgg4);
        scatter_kernel<<<gScat, 256, 0, stream>>>(xu, edge_u2i, agg_i, NEDGE);
        scatter_kernel<<<gScat, 256, 0, stream>>>(xi, edge_i2u, agg_u, NEDGE);
        layer_kernel<<<gI, 256, 0, stream>>>(agg_i, xi, WiL[l], u2i_b,
                                             ln_item_g, ln_item_b, NITEM);
        layer_kernel<<<gU, 256, 0, stream>>>(agg_u, xu, WuL[l], i2u_b,
                                             ln_user_g, ln_user_b, NUSER);
    }

    // ---- 4) emit seed-user embeddings ----
    const int nOut4 = 1024 * CH / 4;
    copy4_kernel<<<(nOut4 + 255) / 256, 256, 0, stream>>>((const float4*)xu,
                                                          (float4*)d_out, nOut4);
}
